// CoarseCorrelationBranch_90159953477954
// MI455X (gfx1250) — matrix operand order repaired and checked by reading, not compiled, not run on hardware
//
#include <hip/hip_runtime.h>
#include <hip/hip_bf16.h>
#include <math.h>

typedef __attribute__((ext_vector_type(16))) __bf16 bf16x16;
typedef __attribute__((ext_vector_type(8)))  float  f32x8;

#define Bn 8
#define Dn 64
#define Hn 256
#define Wn 256
#define Rr 3
#define Kk 49

// d_out layout (floats), outputs concatenated in return order:
// dy, dx, conf_local, conf_global, local_weight, logits
#define NPIX (Bn*Hn*Wn)
#define OFF_DY     0
#define OFF_DX     (NPIX)
#define OFF_CONFL  (2*NPIX)
#define OFF_CONFG  (3*NPIX)
#define OFF_LW     (3*NPIX + Bn)
#define OFF_LOGITS (4*NPIX + Bn)

// ws layout (floats)
#define WS_DYL  0
#define WS_DXL  (NPIX)
#define WS_SUMS (2*NPIX)
#define WS_DYG  (2*NPIX + Bn*Kk)
#define WS_DXG  (2*NPIX + Bn*Kk + Bn)

// ---------------------------------------------------------------------------
// Main fused kernel: normalize -> bf16 LDS tiles -> banded WMMA cost volume
// -> per-pixel softmax (local dy/dx/conf) -> block-partial global sums
// -> coalesced logits writeout.
// Grid: (W/16, H/8, B), block 256 = 8 waves; wave w owns pixel row y0+w.
// ---------------------------------------------------------------------------
__launch_bounds__(256)
__global__ void corr_main(const float* __restrict__ rubin,
                          const float* __restrict__ vis,
                          const float* __restrict__ logt,
                          float* __restrict__ out,
                          float* __restrict__ ws)
{
    __shared__ __attribute__((aligned(32))) __bf16 s_vis[14][22][64]; // 39424 B
    // slot 49 is a trash slot so band extraction can store unconditionally
    __shared__ float s_log[50][8][16];                                // 25600 B

    const int tid  = threadIdx.x;
    const int wv   = tid >> 5;      // wave id = local row
    const int lane = tid & 31;
    const int half = lane >> 4;
    const int m16  = lane & 15;

    const int x0 = blockIdx.x * 16;
    const int y0 = blockIdx.y * 8;
    const int b  = blockIdx.z;

    const size_t planeHW = (size_t)Hn * Wn;
    const float* visb = vis   + (size_t)b * Dn * planeHW;
    const float* rubb = rubin + (size_t)b * Dn * planeHW;

    // ---- cooperative vis tile: load fp32, per-pixel normalize, store bf16 ----
    for (int p = tid; p < 14 * 22; p += 256) {
        int vr = p / 22, vc = p % 22;
        int gy = y0 + vr - Rr; gy = gy < 0 ? 0 : (gy > Hn - 1 ? Hn - 1 : gy); // edge pad
        int gx = x0 + vc - Rr; gx = gx < 0 ? 0 : (gx > Wn - 1 ? Wn - 1 : gx);
        const float* src = visb + (size_t)gy * Wn + gx;
        float v[64]; float ss = 0.f;
#pragma unroll
        for (int ch = 0; ch < 64; ++ch) {
            float x = src[(size_t)ch * planeHW];
            v[ch] = x; ss += x * x;
        }
        float inv = 1.0f / fmaxf(sqrtf(ss), 1e-6f);
#pragma unroll
        for (int ch = 0; ch < 64; ++ch)
            s_vis[vr][vc][ch] = (__bf16)(v[ch] * inv);
    }

    // ---- per-wave rubin row -> A fragments in registers ----
    // lanes L and L+16 share pixel m16; half selects interleaved 8-ch groups
    float r[4][8]; float ss = 0.f;
    {
        const float* src = rubb + (size_t)(y0 + wv) * Wn + (x0 + m16);
        int cs = half * 8;
#pragma unroll
        for (int g = 0; g < 4; ++g)
#pragma unroll
            for (int j = 0; j < 8; ++j) {
                float x = src[(size_t)(g * 16 + cs + j) * planeHW];
                r[g][j] = x; ss += x * x;
            }
    }
    ss += __shfl_xor(ss, 16, 32);                      // combine pixel halves
    float rsq = 0.125f / fmaxf(sqrtf(ss), 1e-6f);      // fold 1/sqrt(D) scale

    bf16x16 afrag[2];
#pragma unroll
    for (int kc = 0; kc < 2; ++kc)
#pragma unroll
        for (int e = 0; e < 8; ++e) {
            afrag[kc][e]     = (__bf16)(r[2 * kc    ][e] * rsq);  // K = kc*32 + cs + e
            afrag[kc][e + 8] = (__bf16)(r[2 * kc + 1][e] * rsq);  // K = kc*32 + 16 + cs + e
        }

    __syncthreads();

    // ---- banded GEMM: per dy, P = A(16x64) * Vrow(64x32); band n-m in [0,6] ----
    float* slog_flat = &s_log[0][0][0];
#pragma unroll
    for (int dy = 0; dy < 7; ++dy) {
        int row = wv + dy;                     // LDS vis row (y0+wv+dy-3)
#pragma unroll
        for (int nt = 0; nt < 2; ++nt) {
            int ng  = nt * 16 + m16;           // output column (vis local col)
            int col = ng > 21 ? 21 : ng;       // cols 22..31 never used
            f32x8 acc = {};
#pragma unroll
            for (int kc = 0; kc < 2; ++kc) {
                const __bf16* vp = &s_vis[row][col][kc * 32 + half * 8];
                bf16x16 bfrag;   // channels kc*32 + half*8 + {0..7, 16..23}
#pragma unroll
                for (int s = 0; s < 16; ++s) bfrag[s] = vp[(s & 7) + (s >> 3) * 16];
                acc = __builtin_amdgcn_wmma_f32_16x16x32_bf16(
                          false, afrag[kc], false, bfrag, (short)0, acc,
                          false, false);
            }
            // extract band entries: unconditional stores, invalid -> trash slot 49
#pragma unroll
            for (int v = 0; v < 8; ++v) {
                int m   = v + half * 8;        // pixel x0+m
                int dxi = ng - m;              // dx + 3
                unsigned k = ((unsigned)dxi < 7u) ? (unsigned)(dy * 7 + dxi) : 49u;
                slog_flat[k * 128 + wv * 16 + m] = acc[v];
            }
        }
    }

    __syncthreads();

    // ---- per-pixel softmax over K=49 -> local dy/dx/conf ----
    float tau  = fmaxf(__expf(logt[0]), 0.001f);
    float itau = 1.0f / tau;
    if (tid < 128) {
        int ry = tid >> 4, cx = tid & 15;
        float l[49]; float mx = -1e30f;
#pragma unroll
        for (int k = 0; k < 49; ++k) { l[k] = s_log[k][ry][cx]; mx = fmaxf(mx, l[k]); }
        float sum = 0.f, dyl = 0.f, dxl = 0.f, pmax = 0.f;
#pragma unroll
        for (int k = 0; k < 49; ++k) {
            float p = __expf((l[k] - mx) * itau);
            sum  += p;
            dyl  += p * (float)(k / 7 - 3);
            dxl  += p * (float)(k % 7 - 3);
            pmax  = fmaxf(pmax, p);
        }
        float is = 1.0f / sum;
        size_t pix = (size_t)b * Hn * Wn + (size_t)(y0 + ry) * Wn + (x0 + cx);
        out[OFF_CONFL + pix] = pmax * is;
        ws[WS_DYL + pix]     = dyl * is;
        ws[WS_DXL + pix]     = dxl * is;
    }

    // ---- block-partial sums for the global mean (one atomic per k) ----
    if (tid < 49) {
        float s = 0.f;
        for (int p = 0; p < 128; ++p) s += s_log[tid][p >> 4][p & 15];
        atomicAdd(&ws[WS_SUMS + b * 49 + tid], s);
    }

    // ---- coalesced logits writeout ----
    for (int i = tid; i < 49 * 8 * 16; i += 256) {
        int k = i >> 7, rem = i & 127, ry = rem >> 4, cx = rem & 15;
        out[OFF_LOGITS + ((size_t)(b * 49 + k) * Hn + (y0 + ry)) * Wn + (x0 + cx)]
            = s_log[k][ry][cx];
    }
}

// ---------------------------------------------------------------------------
// Global softmax over the spatial-mean logits (tiny: B x 49)
// ---------------------------------------------------------------------------
__global__ void corr_global(const float* __restrict__ logt,
                            float* __restrict__ out,
                            float* __restrict__ ws)
{
    int b = blockIdx.x;
    if (threadIdx.x != 0) return;
    float tau = fmaxf(__expf(logt[0]), 0.001f);
    float l[49]; float mx = -1e30f;
    for (int k = 0; k < 49; ++k) {
        l[k] = ws[WS_SUMS + b * 49 + k] * (1.0f / (float)(Hn * Wn));
        mx = fmaxf(mx, l[k]);
    }
    float sum = 0.f, dyg = 0.f, dxg = 0.f, pmax = 0.f;
    for (int k = 0; k < 49; ++k) {
        float p = __expf((l[k] - mx) / tau);
        sum += p;
        dyg += p * (float)(k / 7 - 3);
        dxg += p * (float)(k % 7 - 3);
        pmax = fmaxf(pmax, p);
    }
    out[OFF_CONFG + b] = pmax / sum;
    ws[WS_DYG + b] = dyg / sum;
    ws[WS_DXG + b] = dxg / sum;
}

// ---------------------------------------------------------------------------
// Final blend: dy/dx/local_weight per pixel
// ---------------------------------------------------------------------------
__global__ void corr_combine(float* __restrict__ out, const float* __restrict__ ws)
{
    size_t i = (size_t)blockIdx.x * blockDim.x + threadIdx.x;
    if (i >= (size_t)NPIX) return;
    int b = (int)(i / ((size_t)Hn * Wn));
    float conf = out[OFF_CONFL + i];
    float dyl = ws[WS_DYL + i], dxl = ws[WS_DXL + i];
    float dyg = ws[WS_DYG + b], dxg = ws[WS_DXG + b];
    const float u = 1.0f / 49.0f;
    float lw = (conf - u) / (1.0f - u);
    lw = lw < 0.f ? 0.f : (lw > 1.f ? 1.f : lw);
    out[OFF_DY + i] = lw * dyl + (1.f - lw) * dyg;
    out[OFF_DX + i] = lw * dxl + (1.f - lw) * dxg;
    out[OFF_LW + i] = lw;
}

__global__ void corr_zero(float* __restrict__ ws)
{
    int i = threadIdx.x + blockIdx.x * blockDim.x;
    if (i < Bn * Kk + 2 * Bn) ws[WS_SUMS + i] = 0.f;
}

extern "C" void kernel_launch(void* const* d_in, const int* in_sizes, int n_in,
                              void* d_out, int out_size, void* d_ws, size_t ws_size,
                              hipStream_t stream)
{
    (void)in_sizes; (void)n_in; (void)out_size; (void)ws_size;
    const float* rubin = (const float*)d_in[0];
    const float* vis   = (const float*)d_in[1];
    const float* logt  = (const float*)d_in[2];
    float* out = (float*)d_out;
    float* ws  = (float*)d_ws;

    corr_zero<<<1, 512, 0, stream>>>(ws);
    corr_main<<<dim3(Wn / 16, Hn / 8, Bn), 256, 0, stream>>>(rubin, vis, logt, out, ws);
    corr_global<<<Bn, 32, 0, stream>>>(logt, out, ws);
    corr_combine<<<(NPIX + 255) / 256, 256, 0, stream>>>(out, ws);
}
